// CSNNet_4655744549416
// MI455X (gfx1250) — matrix operand order repaired and checked by reading, not compiled, not run on hardware
//
#include <hip/hip_runtime.h>

typedef __attribute__((ext_vector_type(16))) _Float16 v16h;
typedef __attribute__((ext_vector_type(8)))  float    v8f;

#define BETA   0.9f
#define NSTEP  25
#define BATCH  256
#define L      8192
#define LHALF  4096
#define LIN    32768              // 8 * 4096
#define WPB    4                  // waves per block
#define FEAT_PER_WAVE 512
#define CHUNKS (FEAT_PER_WAVE / 32)

// Wave-local LDS ordering: CDNA5 keeps a wave's DS ops in order (DScnt
// in-order, ISA §7.3), so cross-lane LDS reuse within one wave needs only a
// compiler-ordering fence, not a workgroup barrier.
__device__ __forceinline__ void wave_lds_fence() {
    __builtin_amdgcn_fence(__ATOMIC_ACQ_REL, "wavefront");
}

// ---------------------------------------------------------------------------
// Kernel 0: zero the cur_out accumulation buffer (atomics accumulate into it)
// ---------------------------------------------------------------------------
__global__ void CSNN_zero_kernel(float* co, int n) {
    int i = blockIdx.x * 256 + threadIdx.x;
    if (i < n) co[i] = 0.0f;
}

// ---------------------------------------------------------------------------
// Kernel 1: fused conv1d + maxpool + LIF1 recurrence + WMMA FC partial sums
// grid = (BATCH, 16), block = 128 (4 waves, fully decoupled). Wave w handles
// features [(blockIdx.y*4 + w) * 512, +512) of one batch, 32 at a time:
//   - lane computes cur1 for its feature (conv+pool on the fly, branchless)
//   - 25-step membrane recurrence, f16 spikes -> per-wave LDS tile [t][f]
//   - two v_wmma_f32_16x16x32_f16: D[o, t] += W_chunk(16x32) x Spk(32x16)
// C fragments accumulate over chunks; lanes 0-15 hold rows o=0,1 in c[0],c[1]
// (N = step), flushed with hardware global f32 atomics.
// ---------------------------------------------------------------------------
__global__ __launch_bounds__(128) void CSNN_snn_fc_wmma(
    const float* __restrict__ x, const float* __restrict__ conv_w,
    const float* __restrict__ conv_b, const float* __restrict__ fc_w,
    const float* __restrict__ thr1p, float* __restrict__ curout) {

    __shared__ __align__(128) _Float16 lds_spk[WPB][32][32];   // [wave][t][f_local]

    const int tid  = threadIdx.x;
    const int wave = tid >> 5;
    const int lane = tid & 31;
    const int l15  = lane & 15;
    const int half = lane >> 4;
    const int b    = blockIdx.x;
    const long fbase = (long)(blockIdx.y * WPB + wave) * FEAT_PER_WAVE;
    const float thr1 = thr1p[0];

    // zero the wave's LDS tile once; rows t=25..31 stay zero forever
    {
        unsigned int* p = (unsigned int*)&lds_spk[wave][0][0];
        #pragma unroll
        for (int i = 0; i < 16; ++i) p[lane + i * 32] = 0u;
    }
    wave_lds_fence();

    v8f c0 = {};   // D[o, t] for t = 0..15
    v8f c1 = {};   // D[o, t] for t = 16..31 (25..31 are zero-spike pad)

    for (int ch = 0; ch < CHUNKS; ++ch) {
        const long fchunk = fbase + ch * 32;
        const long f  = fchunk + lane;
        const int  c  = (int)(f >> 12);        // channel (constant per chunk)
        const int  l2 = (int)(f & 4095);       // pooled position

        // ---- conv1d (pad=1) + maxpool(k=2), branchless edge handling ----
        const float w0 = conv_w[c * 3 + 0];
        const float w1 = conv_w[c * 3 + 1];
        const float w2 = conv_w[c * 3 + 2];
        const float bias = conv_b[c];
        const long  xb = (long)b * L + 2 * l2;
        const bool  lo_ok = (l2 > 0);
        const bool  hi_ok = (l2 < LHALF - 1);
        // clamp address into range, mask the value (no EXEC divergence)
        const float xm1 = x[lo_ok ? (xb - 1) : xb] * (lo_ok ? 1.0f : 0.0f);
        const float x0  = x[xb];
        const float x1  = x[xb + 1];
        const float xp2 = x[hi_ok ? (xb + 2) : xb] * (hi_ok ? 1.0f : 0.0f);
        const float ca = xm1 * w0 + x0 * w1 + x1 * w2;
        const float cb = x0  * w0 + x1 * w1 + xp2 * w2;
        const float cur1 = fmaxf(ca, cb) + bias;

        // ---- LIF1: 25-step recurrence, spikes -> LDS [t][f_local] as f16 ----
        float mem = 0.0f;
        #pragma unroll
        for (int t = 0; t < NSTEP; ++t) {
            const float reset = (mem > thr1) ? thr1 : 0.0f;
            mem = BETA * mem + cur1 - reset;
            lds_spk[wave][t][lane] = (mem > thr1) ? (_Float16)1.0f : (_Float16)0.0f;
        }
        wave_lds_fence();

        // ---- A fragment: fc_w chunk as 16x32 f16 (rows = outputs, 2 live) ----
        // A 16x32 in this code's contiguous k order: lanes 0-15: K=0-15; lanes 16-31:
        // K=16-31; element e -> K = half*16+e
        v16h a = {};
        if (l15 < 2) {
            const float* wp = fc_w + (long)l15 * LIN + fchunk;
            #pragma unroll
            for (int e = 0; e < 16; ++e) {
                const int K = half * 16 + e;
                a[e] = (_Float16)wp[K];
            }
        }

        // ---- B fragments from LDS: col = step (l15), rows K = features ----
        // lanes 0-15 hold K rows 0-15, lanes 16-31 rows 16-31 (offset half*16);
        // VGPR v holds rows 2v,2v+1 -> one contiguous 32B read per fragment half
        const _Float16* b0p = &lds_spk[wave][l15][half * 16];
        const _Float16* b1p = &lds_spk[wave][16 + l15][half * 16];
        v16h bf0, bf1;
        #pragma unroll
        for (int e = 0; e < 16; ++e) { bf0[e] = b0p[e]; bf1[e] = b1p[e]; }

        c0 = __builtin_amdgcn_wmma_f32_16x16x32_f16(false, a, false, bf0,
                                                    (short)0, c0, false, false);
        c1 = __builtin_amdgcn_wmma_f32_16x16x32_f16(false, a, false, bf1,
                                                    (short)0, c1, false, false);
        wave_lds_fence();   // B reads complete before next chunk's spike stores
    }

    // ---- flush partial cur_out[t][b][o] with hardware f32 atomics ----
    if (lane < 16) {
        float* dst0 = &curout[(lane * BATCH + b) * 2];
        __hip_atomic_fetch_add(&dst0[0], c0[0], __ATOMIC_RELAXED, __HIP_MEMORY_SCOPE_AGENT);
        __hip_atomic_fetch_add(&dst0[1], c0[1], __ATOMIC_RELAXED, __HIP_MEMORY_SCOPE_AGENT);
        const int t1 = 16 + lane;
        if (t1 < NSTEP) {
            float* dst1 = &curout[(t1 * BATCH + b) * 2];
            __hip_atomic_fetch_add(&dst1[0], c1[0], __ATOMIC_RELAXED, __HIP_MEMORY_SCOPE_AGENT);
            __hip_atomic_fetch_add(&dst1[1], c1[1], __ATOMIC_RELAXED, __HIP_MEMORY_SCOPE_AGENT);
        }
    }
}

// ---------------------------------------------------------------------------
// Kernel 2: output-LIF recurrence over [B,2]; writes spk_rec then mem_rec
// ---------------------------------------------------------------------------
__global__ void CSNN_memout_kernel(const float* __restrict__ curout,
                                   const float* __restrict__ fc_b,
                                   const float* __restrict__ thr_out_p,
                                   float* __restrict__ out) {
    const int i = blockIdx.x * blockDim.x + threadIdx.x;
    if (i >= BATCH * 2) return;
    const int b = i >> 1, o = i & 1;
    const float bias = fc_b[o];
    const float th = thr_out_p[0];
    float mem = 0.0f;
    for (int t = 0; t < NSTEP; ++t) {
        const float cur = curout[(t * BATCH + b) * 2 + o] + bias;
        const float reset = (mem > th) ? th : 0.0f;
        mem = BETA * mem + cur - reset;
        const float spk = (mem > th) ? 1.0f : 0.0f;
        out[(t * BATCH + b) * 2 + o] = spk;                       // spk_rec
        out[NSTEP * BATCH * 2 + (t * BATCH + b) * 2 + o] = mem;   // mem_rec
    }
}

// ---------------------------------------------------------------------------
extern "C" void kernel_launch(void* const* d_in, const int* in_sizes, int n_in,
                              void* d_out, int out_size, void* d_ws, size_t ws_size,
                              hipStream_t stream) {
    const float* x       = (const float*)d_in[0];
    const float* conv_w  = (const float*)d_in[1];
    const float* conv_b  = (const float*)d_in[2];
    const float* fc_w    = (const float*)d_in[3];
    const float* fc_b    = (const float*)d_in[4];
    const float* thr1    = (const float*)d_in[5];
    const float* thr_out = (const float*)d_in[6];
    float* curout = (float*)d_ws;          // 25*256*2 floats = 50 KB scratch
    float* out    = (float*)d_out;

    const int ncur = NSTEP * BATCH * 2;
    CSNN_zero_kernel<<<dim3((ncur + 255) / 256), dim3(256), 0, stream>>>(curout, ncur);
    CSNN_snn_fc_wmma<<<dim3(BATCH, 16), dim3(128), 0, stream>>>(
        x, conv_w, conv_b, fc_w, thr1, curout);
    CSNN_memout_kernel<<<dim3(2), dim3(256), 0, stream>>>(curout, fc_b, thr_out, out);
}
